// GCN_12850542150061
// MI455X (gfx1250) — matrix operand order repaired and checked by reading, not compiled, not run on hardware
//
#include <hip/hip_runtime.h>

typedef __attribute__((ext_vector_type(16))) __bf16 v16bf;
typedef __attribute__((ext_vector_type(8)))  __bf16 v8bf;
typedef __attribute__((ext_vector_type(8)))  float  v8f;

// ---------------------------------------------------------------------------
// GEMM: out[M,N] = A[M,K] (f32) x W[K,N] (f32), computed in bf16 WMMA with
// f32 accumulation. Block = 256 threads (8 wave32), block tile = 64 x N.
// LDS: A tile as bf16 [64][K], W transposed as bf16 [N][K].
// ---------------------------------------------------------------------------
template<int K, int N>
__global__ __launch_bounds__(256)
void gemm_bf16_wmma(const float* __restrict__ A, const float* __restrict__ W,
                    float* __restrict__ out, int Mtot) {
  constexpr int CT  = N / 16;   // column tiles total
  constexpr int CT2 = CT / 2;   // column tiles per wave
  extern __shared__ __align__(32) __bf16 smem[];
  __bf16* sA  = smem;           // [64][K]
  __bf16* sWt = smem + 64 * K;  // [N][K]  (transposed W)

  const int tid  = threadIdx.x;
  const int row0 = blockIdx.x * 64;

  // Stage W transposed (f32 -> bf16): column n of W becomes contiguous in LDS.
  for (int i = tid; i < K * N; i += 256) {
    int k = i / N, n = i % N;
    sWt[n * K + k] = (__bf16)W[i];
  }
  // Stage 64 x K A tile (f32 -> bf16), vectorized 4-wide, zero-pad past Mtot.
  for (int i = tid * 4; i < 64 * K; i += 256 * 4) {
    int r = i / K, k = i % K;
    int gr = row0 + r;
    float4 v = make_float4(0.f, 0.f, 0.f, 0.f);
    if (gr < Mtot) v = *(const float4*)(A + (size_t)gr * K + k);
    __bf16* s = sA + r * K + k;
    s[0] = (__bf16)v.x; s[1] = (__bf16)v.y; s[2] = (__bf16)v.z; s[3] = (__bf16)v.w;
  }
  __syncthreads();

  const int wave = tid >> 5;
  const int lane = tid & 31;
  const int half = lane >> 4;   // selects K sub-group per ISA 16-bit layouts
  const int idx  = lane & 15;   // A: row, B/C/D: column
  const int rowTile  = wave & 3;
  const int colTile0 = (wave >> 2) * CT2;

  v8f vzero = {};
  v8f acc[CT2];
  #pragma unroll
  for (int t = 0; t < CT2; ++t) acc[t] = vzero;

  const __bf16* aRow = sA + (rowTile * 16 + idx) * K;

  for (int k0 = 0; k0 < K; k0 += 32) {
    // A fragment 16x32: lane half 0 -> K = k0+{0..15},
    //                   lane half 1 -> K = k0+{16..31}
    v8bf alo = *(const v8bf*)(aRow + k0 + half * 16);
    v8bf ahi = *(const v8bf*)(aRow + k0 + half * 16 + 8);
    v16bf a;
    #pragma unroll
    for (int j = 0; j < 8; ++j) { a[j] = alo[j]; a[j + 8] = ahi[j]; }

    #pragma unroll
    for (int t = 0; t < CT2; ++t) {
      // B fragment 32x16: column = colTile*16 + idx, K = k0 + half*16 + j,
      // contiguous in transposed LDS -> one 32B load.
      const __bf16* bp = sWt + ((colTile0 + t) * 16 + idx) * K + k0 + half * 16;
      v16bf b = *(const v16bf*)bp;
      acc[t] = __builtin_amdgcn_wmma_f32_16x16x32_bf16(
          /*neg_a=*/false, a, /*neg_b=*/false, b,
          /*c_mod=*/(short)0, acc[t], /*reuse_a=*/false, /*reuse_b=*/false);
    }
  }

  // C/D layout: VGPR r, lanes 0-15 -> M=r ; lanes 16-31 -> M=8+r ; N=idx.
  #pragma unroll
  for (int t = 0; t < CT2; ++t) {
    const int colBase = (colTile0 + t) * 16;
    #pragma unroll
    for (int r = 0; r < 8; ++r) {
      int gr = row0 + rowTile * 16 + half * 8 + r;
      if (gr < Mtot) out[(size_t)gr * N + colBase + idx] = acc[t][r];
    }
  }
}

// ---------------------------------------------------------------------------
// SpMM scatter: out[row[e], :] += val[e] * feat[col[e], :]   (COO edges)
// One wave32 per edge, D/32 features per lane, f32 atomic adds (L2-resident).
// ---------------------------------------------------------------------------
template<int D>
__global__ __launch_bounds__(256)
void spmm_scatter(const float* __restrict__ feat, const float* __restrict__ val,
                  const int* __restrict__ erow, const int* __restrict__ ecol,
                  float* __restrict__ out, int E) {
  int e = blockIdx.x * 8 + (threadIdx.x >> 5);
  if (e >= E) return;
  const int lane = threadIdx.x & 31;
  constexpr int P = D / 32;
  const float v = val[e];
  const float* src = feat + (size_t)ecol[e] * D + lane * P;
  float*       dst = out  + (size_t)erow[e] * D + lane * P;
  #pragma unroll
  for (int j = 0; j < P; ++j)
    __hip_atomic_fetch_add(dst + j, v * src[j],
                           __ATOMIC_RELAXED, __HIP_MEMORY_SCOPE_AGENT);
}

__global__ void fill_zero(float* __restrict__ p, int n) {
  int i = blockIdx.x * blockDim.x + threadIdx.x;
  if (i < n) p[i] = 0.f;
}

__global__ void fill_bias(float* __restrict__ p, const float* __restrict__ b,
                          int n, int mask) {
  int i = blockIdx.x * blockDim.x + threadIdx.x;
  if (i < n) p[i] = b[i & mask];
}

__global__ void bias_relu(float* __restrict__ h, const float* __restrict__ b,
                          int n, int mask) {
  int i = blockIdx.x * blockDim.x + threadIdx.x;
  if (i < n) {
    float v = h[i] + b[i & mask];
    h[i] = v > 0.f ? v : 0.f;
  }
}

extern "C" void kernel_launch(void* const* d_in, const int* in_sizes, int n_in,
                              void* d_out, int out_size, void* d_ws, size_t ws_size,
                              hipStream_t stream) {
  (void)n_in; (void)out_size; (void)ws_size;
  constexpr int NF = 512, NH = 128, NC = 64;

  const float* x    = (const float*)d_in[0];
  const float* W1   = (const float*)d_in[1];
  const float* b1   = (const float*)d_in[2];
  const float* W2   = (const float*)d_in[3];
  const float* b2   = (const float*)d_in[4];
  const float* eval_= (const float*)d_in[5];
  const int*   erow = (const int*)d_in[6];
  const int*   ecol = (const int*)d_in[7];
  float* out = (float*)d_out;

  const int M = in_sizes[0] / NF;   // 50000 nodes
  const int E = in_sizes[5];        // 800000 edges

  float* XW = (float*)d_ws;             // [M, NH]
  float* H  = XW + (size_t)M * NH;      // [M, NH]
  float* HW = XW;                       // [M, NC]  (reuse XW region)

  const dim3 blk(256);
  const int  mb = (M + 63) / 64;

  // 1) XW = X @ W1   (bf16 WMMA, f32 accum)
  size_t lds1 = (size_t)(64 * NF + NF * NH) * sizeof(__bf16);   // 192 KB
  gemm_bf16_wmma<NF, NH><<<mb, blk, lds1, stream>>>(x, W1, XW, M);

  // 2) H = segment_sum(val * XW[col])   then   H = relu(H + b1)
  fill_zero<<<((size_t)M * NH + 255) / 256, blk, 0, stream>>>(H, M * NH);
  spmm_scatter<NH><<<(E + 7) / 8, blk, 0, stream>>>(XW, eval_, erow, ecol, H, E);
  bias_relu<<<((size_t)M * NH + 255) / 256, blk, 0, stream>>>(H, b1, M * NH, NH - 1);

  // 3) HW = H @ W2
  size_t lds2 = (size_t)(64 * NH + NH * NC) * sizeof(__bf16);   // 32 KB
  gemm_bf16_wmma<NH, NC><<<mb, blk, lds2, stream>>>(H, W2, HW, M);

  // 4) out = b2 + segment_sum(val * HW[col])
  fill_bias<<<((size_t)M * NC + 255) / 256, blk, 0, stream>>>(out, b2, M * NC, NC - 1);
  spmm_scatter<NC><<<(E + 7) / 8, blk, 0, stream>>>(HW, eval_, erow, ecol, out, E);
}
